// DecoupledAttentionAggregation_46901042872938
// MI455X (gfx1250) — compile-verified, matrix operand order checked by reading, not run on hardware
//
#include <hip/hip_runtime.h>
#include <hip/hip_bf16.h>

// CDNA5 (gfx1250) wave32 WMMA types
typedef __attribute__((ext_vector_type(16))) __bf16 v16bf;
typedef __attribute__((ext_vector_type(8)))  __bf16 v8bf;
typedef __attribute__((ext_vector_type(8)))  float  v8f;
typedef __attribute__((ext_vector_type(4)))  float  v4f;

#define TOPK 10

static __device__ __forceinline__ v16bf cat8(v8bf lo, v8bf hi) {
  return __builtin_shufflevector(lo, hi, 0, 1, 2, 3, 4, 5, 6, 7,
                                 8, 9, 10, 11, 12, 13, 14, 15);
}

// ---- order-preserving float <-> uint key for atomic segment-max ----
__device__ __forceinline__ unsigned fkey(float f) {
  unsigned b = __float_as_uint(f);
  return (b & 0x80000000u) ? ~b : (b | 0x80000000u);
}
__device__ __forceinline__ float funkey(unsigned k) {
  unsigned b = (k & 0x80000000u) ? (k ^ 0x80000000u) : ~k;
  return __uint_as_float(b);
}

// =====================================================================
// init kernels
// =====================================================================
__global__ void k_zero_v4(v4f* __restrict__ p, int n4) {
  int i = blockIdx.x * 256 + threadIdx.x;
  if (i < n4) { v4f z = {0.f, 0.f, 0.f, 0.f}; p[i] = z; }
}

__global__ void k_zero_f32(float* __restrict__ p, int n) {
  int i = blockIdx.x * 256 + threadIdx.x;
  if (i < n) p[i] = 0.0f;
}

__global__ void k_init_nodes(unsigned* __restrict__ smax, float* __restrict__ den,
                             int* __restrict__ deg, int n) {
  int i = blockIdx.x * 256 + threadIdx.x;
  if (i < n) { smax[i] = 0u; den[i] = 0.0f; deg[i] = 0; }
}

// =====================================================================
// weight prep: transpose + f32->bf16 so WMMA B-operand loads are two
// contiguous b128 loads per lane.
//   whT[n*64 + k] = bf16(whW[k*64 + n])   (64x64)
//   weT[n*32 + k] = bf16(weW[k*64 + n])   (32x64 -> 64x32)
// =====================================================================
__global__ void k_prep_w(const float* __restrict__ whW, const float* __restrict__ weW,
                         __bf16* __restrict__ whT, __bf16* __restrict__ weT) {
  int i = blockIdx.x * 256 + threadIdx.x;
  if (i < 64 * 64) {
    const int n = i >> 6, k = i & 63;
    whT[n * 64 + k] = (__bf16)whW[k * 64 + n];
  }
  if (i < 32 * 64) {
    const int n = i >> 5, k = i & 31;
    weT[n * 32 + k] = (__bf16)weW[k * 64 + n];
  }
}

// =====================================================================
// Wh = h @ whW   [N,64] x [64,64] -> [N,64]
// one 16x16 output tile per wave; K=64 as two v_wmma_f32_16x16x32_bf16.
// A loads: 4x b128 f32 per ktile; B loads: 2x b128 packed bf16 per ktile.
// =====================================================================
__global__ __launch_bounds__(128) void k_gemm_wh(
    const float* __restrict__ h, const __bf16* __restrict__ whT,
    float* __restrict__ Wh, int n) {
  const int lane = threadIdx.x & 31;
  const int wid  = threadIdx.x >> 5;                 // output col tile 0..3
  const int l15  = lane & 15;
  const int kset = (lane >> 4) * 8;
  const int ncol = wid * 16 + l15;
  int mrow = blockIdx.x * 16 + l15;
  if (mrow >= n) mrow = n - 1;                        // clamp: keep EXEC all-1s

  const float*  arow = h + (size_t)mrow * 64;
  const __bf16* bcol = whT + ncol * 64;

  v8f c = {};
#pragma unroll
  for (int kt = 0; kt < 2; ++kt) {
    const int k0 = kt * 32;
    const v4f a0 = *(const v4f*)(arow + k0 + kset);
    const v4f a1 = *(const v4f*)(arow + k0 + kset + 4);
    const v4f a2 = *(const v4f*)(arow + k0 + 16 + kset);
    const v4f a3 = *(const v4f*)(arow + k0 + 16 + kset + 4);
    v16bf a;
#pragma unroll
    for (int j = 0; j < 4; ++j) {
      a[j]      = (__bf16)a0[j];
      a[j + 4]  = (__bf16)a1[j];
      a[j + 8]  = (__bf16)a2[j];
      a[j + 12] = (__bf16)a3[j];
    }
    const v8bf blo = *(const v8bf*)(bcol + k0 + kset);
    const v8bf bhi = *(const v8bf*)(bcol + k0 + 16 + kset);
    const v16bf b = cat8(blo, bhi);
    c = __builtin_amdgcn_wmma_f32_16x16x32_bf16(false, a, false, b,
                                                (short)0, c, false, false);
  }

  const int mb   = (lane >> 4) * 8;                   // C/D: lanes16-31 hold M=8..15
  const int base = blockIdx.x * 16 + mb;
  if (blockIdx.x * 16 + 16 <= n) {                    // uniform full-tile fast path
#pragma unroll
    for (int r = 0; r < 8; ++r) Wh[(size_t)(base + r) * 64 + ncol] = c[r];
  } else {
#pragma unroll
    for (int r = 0; r < 8; ++r)
      if (base + r < n) Wh[(size_t)(base + r) * 64 + ncol] = c[r];
  }
}

// =====================================================================
// per-node attention projections
// =====================================================================
__global__ void k_nodeproj(const float* __restrict__ h, const float* __restrict__ aw,
                           float* __restrict__ srow, float* __restrict__ scol, int n) {
  int i = blockIdx.x * 256 + threadIdx.x;
  if (i >= n) return;
  const float* x = h + (size_t)i * 64;
  float r0 = 0, r1 = 0, r2 = 0, r3 = 0, c0 = 0, c1 = 0, c2 = 0, c3 = 0;
#pragma unroll 4
  for (int k = 0; k < 64; ++k) {
    const float xv = x[k];
    const float* wr = aw + k * 4;
    const float* wc = aw + (64 + k) * 4;
    r0 += xv * wr[0]; r1 += xv * wr[1]; r2 += xv * wr[2]; r3 += xv * wr[3];
    c0 += xv * wc[0]; c1 += xv * wc[1]; c2 += xv * wc[2]; c3 += xv * wc[3];
  }
  srow[i * 4 + 0] = r0; srow[i * 4 + 1] = r1; srow[i * 4 + 2] = r2; srow[i * 4 + 3] = r3;
  scol[i * 4 + 0] = c0; scol[i * 4 + 1] = c1; scol[i * 4 + 2] = c2; scol[i * 4 + 3] = c3;
}

// =====================================================================
// per-edge score (leaky-relu per head, mean over 4 heads), seg-max, degree
// =====================================================================
__global__ void k_scores(const float* __restrict__ ea, const float* __restrict__ aw,
                         const float* __restrict__ srow, const float* __restrict__ scol,
                         const int* __restrict__ row, const int* __restrict__ col,
                         float* __restrict__ scoreBuf, unsigned* __restrict__ smax,
                         int* __restrict__ deg, int E) {
  int e = blockIdx.x * 256 + threadIdx.x;
  if (e >= E) return;
  const float* a3 = aw + 128 * 4;
  const float* x  = ea + (size_t)e * 32;
  float s0 = 0, s1 = 0, s2 = 0, s3 = 0;
#pragma unroll 8
  for (int k = 0; k < 32; ++k) {
    const float xv = x[k];
    const float* w = a3 + k * 4;
    s0 += xv * w[0]; s1 += xv * w[1]; s2 += xv * w[2]; s3 += xv * w[3];
  }
  const int r = row[e], c = col[e];
  float v[4];
  v[0] = s0 + srow[r * 4 + 0] + scol[c * 4 + 0];
  v[1] = s1 + srow[r * 4 + 1] + scol[c * 4 + 1];
  v[2] = s2 + srow[r * 4 + 2] + scol[c * 4 + 2];
  v[3] = s3 + srow[r * 4 + 3] + scol[c * 4 + 3];
  float s = 0;
#pragma unroll
  for (int hh = 0; hh < 4; ++hh) {
    const float t = v[hh];
    s += (t >= 0.0f) ? t : 0.2f * t;
  }
  s *= 0.25f;
  scoreBuf[e] = s;
  atomicMax(&smax[c], fkey(s));
  atomicAdd(&deg[c], 1);
}

// =====================================================================
// ex = exp(score - segmax); den = segment_sum(ex)   (in-place on score buf)
// =====================================================================
__global__ void k_expden(const int* __restrict__ col, const unsigned* __restrict__ smax,
                         float* __restrict__ ex, float* __restrict__ den, int E) {
  int e = blockIdx.x * 256 + threadIdx.x;
  if (e >= E) return;
  const int c = col[e];
  const float x = __expf(ex[e] - funkey(smax[c]));
  ex[e] = x;
  atomicAdd(&den[c], x);
}

// =====================================================================
// CSR build: block scan of deg -> exclusive start offsets
// =====================================================================
__global__ void k_scan_block(const int* __restrict__ deg, int* __restrict__ startArr,
                             int* __restrict__ bsum, int n) {
  __shared__ int sm[256];
  const int t = threadIdx.x;
  const int i = blockIdx.x * 256 + t;
  const int v = (i < n) ? deg[i] : 0;
  sm[t] = v;
  __syncthreads();
  for (int off = 1; off < 256; off <<= 1) {
    const int x = (t >= off) ? sm[t - off] : 0;
    __syncthreads();
    sm[t] += x;
    __syncthreads();
  }
  if (i < n) startArr[i] = sm[t] - v;                 // exclusive prefix
  if (t == 255) bsum[blockIdx.x] = sm[255];
}

__global__ void k_scan_sums(int* __restrict__ bsum, int nb) {
  if (threadIdx.x != 0 || blockIdx.x != 0) return;
  int running = 0;
  for (int b = 0; b < nb; ++b) { const int t = bsum[b]; bsum[b] = running; running += t; }
}

__global__ void k_scan_add(int* __restrict__ startArr, int* __restrict__ cursor,
                           const int* __restrict__ bsum, int n) {
  int i = blockIdx.x * 256 + threadIdx.x;
  if (i >= n) return;
  const int s = startArr[i] + bsum[blockIdx.x];
  startArr[i] = s;
  cursor[i]   = s;
}

__global__ void k_scatter(const int* __restrict__ col, int* __restrict__ cursor,
                          int* __restrict__ csr, int E) {
  int e = blockIdx.x * 256 + threadIdx.x;
  if (e >= E) return;
  const int p = atomicAdd(&cursor[col[e]], 1);
  csr[p] = e;
}

// =====================================================================
// per-node top-k: rank edges by (ex desc, edge idx asc); attn = keep*ex/den
// =====================================================================
__global__ void k_topk(const int* __restrict__ startArr, const int* __restrict__ deg,
                       const int* __restrict__ csr, const float* __restrict__ ex,
                       const float* __restrict__ den, float* __restrict__ attn, int n) {
  int node = blockIdx.x * 256 + threadIdx.x;
  if (node >= n) return;
  const int s = startArr[node];
  const int d = deg[node];
  const float dn = den[node];
  const float inv = (dn != 0.0f) ? (1.0f / dn) : 0.0f;
  for (int i = 0; i < d; ++i) {
    const int ei = csr[s + i];
    const float vi = ex[ei];
    int rank = 0;
    for (int j = 0; j < d; ++j) {
      const int ej = csr[s + j];
      const float vj = ex[ej];
      if (vj > vi || (vj == vi && ej < ei)) ++rank;
    }
    attn[ei] = (rank < TOPK) ? vi * inv : 0.0f;
  }
}

// =====================================================================
// fused edge message kernel (WMMA):
//   We = edge_attr @ weW (16-edge x 16-col tile per wave, K=32 = one WMMA)
//   msg = relu(We + Wh[row] + (whb+web)); out[col, slice*64+n] += attn*msg
// Edge scalars preloaded + Wh rows prefetched before the WMMA so gather
// latency overlaps the matrix op.
// =====================================================================
__global__ __launch_bounds__(128) void k_edge_msg(
    const float* __restrict__ ea, const __bf16* __restrict__ weT,
    const float* __restrict__ whb, const float* __restrict__ web,
    const float* __restrict__ Wh, const float* __restrict__ attn,
    const int* __restrict__ row, const int* __restrict__ col,
    const int* __restrict__ labels, float* __restrict__ out, int E) {
  const int lane  = threadIdx.x & 31;
  const int wid   = threadIdx.x >> 5;                // output col tile 0..3
  const int l15   = lane & 15;
  const int kset  = (lane >> 4) * 8;
  const int etile = blockIdx.x * 16;
  const int ncol  = wid * 16 + l15;
  const int mb    = (lane >> 4) * 8;
  int me = etile + l15;
  if (me >= E) me = E - 1;                            // clamp: EXEC stays all-1s

  // --- preload per-edge scalars; prefetch gathered Wh rows ---
  int   rs[8], cs[8];
  float wAtt[8];
#pragma unroll
  for (int r = 0; r < 8; ++r) {
    int e = etile + mb + r;
    if (e >= E) e = E - 1;
    wAtt[r] = attn[e];
    rs[r]   = row[e];
    cs[r]   = col[e];
    __builtin_prefetch(Wh + (size_t)rs[r] * 64 + ncol, 0, 1);
  }

  // --- A: 4x b128 f32 loads; B: 2x b128 packed-bf16 loads ---
  const float* arow = ea + (size_t)me * 32;
  const v4f a0 = *(const v4f*)(arow + kset);
  const v4f a1 = *(const v4f*)(arow + kset + 4);
  const v4f a2 = *(const v4f*)(arow + 16 + kset);
  const v4f a3 = *(const v4f*)(arow + 16 + kset + 4);
  v16bf a;
#pragma unroll
  for (int j = 0; j < 4; ++j) {
    a[j]      = (__bf16)a0[j];
    a[j + 4]  = (__bf16)a1[j];
    a[j + 8]  = (__bf16)a2[j];
    a[j + 12] = (__bf16)a3[j];
  }
  const __bf16* bcol = weT + ncol * 32;
  const v16bf b = cat8(*(const v8bf*)(bcol + kset),
                       *(const v8bf*)(bcol + 16 + kset));
  v8f c = {};
  c = __builtin_amdgcn_wmma_f32_16x16x32_bf16(false, a, false, b,
                                              (short)0, c, false, false);

  const float bias = whb[ncol] + web[ncol];
#pragma unroll
  for (int r = 0; r < 8; ++r) {
    const int e = etile + mb + r;
    if (e < E) {
      const float w = wAtt[r];
      if (w != 0.0f) {
        float v = c[r] + Wh[(size_t)rs[r] * 64 + ncol] + bias;
        v = (v > 0.0f) ? v : 0.0f;
        const int lr = labels[rs[r]], lc = labels[cs[r]];
        const int slice = (lr == -1 || lc == -1) ? 2 : ((lr == lc) ? 0 : 1);
        atomicAdd(out + (size_t)cs[r] * 192 + slice * 64 + ncol, v * w);
      }
    }
  }
}

// =====================================================================
// host launcher
// =====================================================================
extern "C" void kernel_launch(void* const* d_in, const int* in_sizes, int n_in,
                              void* d_out, int out_size, void* d_ws, size_t ws_size,
                              hipStream_t stream) {
  const float* h      = (const float*)d_in[0];
  const int*   eidx   = (const int*)  d_in[1];
  const float* eattr  = (const float*)d_in[2];
  const int*   labels = (const int*)  d_in[3];
  const float* aw     = (const float*)d_in[4];
  const float* whW    = (const float*)d_in[5];
  const float* whb    = (const float*)d_in[6];
  const float* weW    = (const float*)d_in[7];
  const float* web    = (const float*)d_in[8];
  float* out = (float*)d_out;

  const int N = in_sizes[0] / 64;       // hidden dim 64
  const int E = in_sizes[1] / 2;        // edge_index is [2,E]
  const int* row = eidx;
  const int* col = eidx + E;

  // ---- carve workspace (~56 MB) ----
  char* base = (char*)d_ws;
  size_t off = 0;
  auto carve = [&](size_t bytes) -> void* {
    off = (off + 255) & ~(size_t)255;
    void* p = base + off;
    off += bytes;
    return p;
  };
  float*    Wh     = (float*)   carve((size_t)N * 64 * 4);
  float*    srow   = (float*)   carve((size_t)N * 4 * 4);
  float*    scol   = (float*)   carve((size_t)N * 4 * 4);
  unsigned* smax   = (unsigned*)carve((size_t)N * 4);
  float*    den    = (float*)   carve((size_t)N * 4);
  int*      deg    = (int*)     carve((size_t)N * 4);
  int*      startA = (int*)     carve((size_t)N * 4);
  int*      cursor = (int*)     carve((size_t)N * 4);
  const int NB = (N + 255) / 256;
  int*      bsum   = (int*)     carve((size_t)NB * 4);
  float*    ex     = (float*)   carve((size_t)E * 4);
  float*    attn   = (float*)   carve((size_t)E * 4);
  int*      csr    = (int*)     carve((size_t)E * 4);
  __bf16*   whT    = (__bf16*)  carve((size_t)64 * 64 * 2);
  __bf16*   weT    = (__bf16*)  carve((size_t)64 * 32 * 2);
  (void)ws_size; (void)n_in;

  const int EB = (E + 255) / 256;

  // 0) zero output (atomically accumulated; harness poisons it) + node stats
  const int n4 = out_size >> 2;
  if (n4 > 0) k_zero_v4<<<(n4 + 255) / 256, 256, 0, stream>>>((v4f*)out, n4);
  if (out_size & 3)
    k_zero_f32<<<1, 256, 0, stream>>>(out + (n4 << 2), out_size & 3);
  k_init_nodes<<<NB, 256, 0, stream>>>(smax, den, deg, N);

  // 0b) weight transpose + bf16 pre-conversion (12 KB, trivial)
  k_prep_w<<<16, 256, 0, stream>>>(whW, weW, whT, weT);

  // 1) Wh = h @ whW  (WMMA)
  k_gemm_wh<<<(N + 15) / 16, 128, 0, stream>>>(h, whT, Wh, N);

  // 2) per-node attention projections
  k_nodeproj<<<NB, 256, 0, stream>>>(h, aw, srow, scol, N);

  // 3) per-edge scores + segment max + degree
  k_scores<<<EB, 256, 0, stream>>>(eattr, aw, srow, scol, row, col, ex, smax, deg, E);

  // 4) exp + denominator (in-place on ex)
  k_expden<<<EB, 256, 0, stream>>>(col, smax, ex, den, E);

  // 5) CSR: scan degrees, scatter edge ids
  k_scan_block<<<NB, 256, 0, stream>>>(deg, startA, bsum, N);
  k_scan_sums<<<1, 1, 0, stream>>>(bsum, NB);
  k_scan_add<<<NB, 256, 0, stream>>>(startA, cursor, bsum, N);
  k_scatter<<<EB, 256, 0, stream>>>(col, cursor, csr, E);

  // 6) per-destination top-k pruning -> final attn weights
  k_topk<<<NB, 256, 0, stream>>>(startA, deg, csr, ex, den, attn, N);

  // 7) fused WMMA message + masked segment-sum aggregation
  k_edge_msg<<<(E + 15) / 16, 128, 0, stream>>>(eattr, weT, whb, web, Wh, attn,
                                                row, col, labels, out, E);
}
